// RaggedHSTUAttn_49065706389786
// MI455X (gfx1250) — matrix operand order repaired and checked by reading, not compiled, not run on hardware
//
#include <hip/hip_runtime.h>
#include <math.h>

// ---------------- problem constants (from reference) ----------------
#define ZZ 128
#define NMAX 512
#define HH 4
#define DD 64
#define NUM_BUCKETS 2048
#define ALPHA_F 0.08838834764831843f
#define INV_N (1.0f / 512.0f)
#define LOG2E 1.4426950408889634f
#define MASK_NEG (-1.0e30f)

typedef __bf16 bf16;
typedef __attribute__((ext_vector_type(16))) __bf16 v16bf;
typedef __attribute__((ext_vector_type(8)))  __bf16 v8bf;
typedef __attribute__((ext_vector_type(4)))  __bf16 v4bf;
typedef __attribute__((ext_vector_type(8)))  float  v8f;
typedef __attribute__((ext_vector_type(4)))  float  v4f;

#define KSTR 72   // k_lds row stride (bf16), 64-wide rows, padded
#define VSTR 40   // vT_lds row stride (bf16), 32-wide rows, padded
#define SSTR 40   // s_lds row stride (bf16), 32-wide rows, padded
#define BSTR 36   // bias_lds row stride (f32), 32-wide rows, padded

__device__ inline v16bf cat16(v8bf lo, v8bf hi) {
  return __builtin_shufflevector(lo, hi, 0,1,2,3,4,5,6,7,8,9,10,11,12,13,14,15);
}
__device__ inline v16bf lds_load16(const bf16* p) {   // 16 contiguous bf16 (32B)
  v8bf lo = *reinterpret_cast<const v8bf*>(p);
  v8bf hi = *reinterpret_cast<const v8bf*>(p + 8);
  return cat16(lo, hi);
}
__device__ inline v8bf load_cvt8_scaled(const float* p, float s) {
  v8f f = *reinterpret_cast<const v8f*>(p);
  v8bf r;
#pragma unroll
  for (int i = 0; i < 8; ++i) r[i] = (__bf16)(f[i] * s);
  return r;
}

// silu without the 1/N scale (folded into epilogue).
__device__ inline float silu_fast(float x) {
#if __has_builtin(__builtin_amdgcn_tanhf)
  float h = 0.5f * x;
  float t = __builtin_amdgcn_tanhf(h);   // v_tanh_f32 (1 TRANS)
  return __builtin_fmaf(h, t, h);        // x*sigmoid(x), fuses to v_fma_mix*_bf16
#else
  float e = __builtin_amdgcn_exp2f(x * (-LOG2E));
  float r = __builtin_amdgcn_rcpf(1.0f + e);
  return x * r;
#endif
}

// =====================================================================
// Kernel 1: head-invariant bias pre-pass.
// bias[z][m][n] = tw[bucket] + pw[n-m+511], masked (n>m or n>=len) -> -1e30
// Only lower block-triangle tiles (jt <= 4*mb+3) are written = exactly the
// tiles the attention kernel reads.
// =====================================================================
__global__ __launch_bounds__(256) void hstu_bias_pre(
    const float* __restrict__ tw, const float* __restrict__ pw,
    const int* __restrict__ seq_offsets, const int* __restrict__ timestamps,
    bf16* __restrict__ bias)
{
  __shared__ float ts_lds[NMAX + 1];

  int q = blockIdx.x % 40;               // 40 tiles per z (4+8+12+16)
  const int z = blockIdx.x / 40;
  int mb, jt;
  if (q < 4)       { mb = 0; jt = q; }
  else if (q < 12) { mb = 1; jt = q - 4; }
  else if (q < 24) { mb = 2; jt = q - 12; }
  else             { mb = 3; jt = q - 24; }

  const int len = seq_offsets[z + 1] - seq_offsets[z];
  const int tid = threadIdx.x;

  for (int i = tid; i <= NMAX; i += 256)
    ts_lds[i] = (float)timestamps[z * (NMAX + 1) + i];
  __syncthreads();

  const int m_local = tid >> 1;
  const int half    = tid & 1;
  const int m = mb * 128 + m_local;
  const int n0 = jt * 32 + half * 16;
  const float ts_m = ts_lds[m + 1];

  v8bf outv[2];
#pragma unroll
  for (int i = 0; i < 16; ++i) {
    int n = n0 + i;
    float dt = ts_m - ts_lds[n];
    float bt = __builtin_amdgcn_sqrtf(fmaxf(dt * (1.0f / 60.0f), 0.0f));
    int bucket = min((int)bt, NUM_BUCKETS);
    float b = tw[bucket] + pw[n - m + (NMAX - 1)];
    bool ok = (n <= m) && (n < len);
    outv[i >> 3][i & 7] = (__bf16)(ok ? b : MASK_NEG);
  }
  bf16* dst = bias + (size_t)z * (NMAX * NMAX) + (size_t)m * NMAX + n0;
  *reinterpret_cast<v8bf*>(dst)     = outv[0];
  *reinterpret_cast<v8bf*>(dst + 8) = outv[1];
}

// =====================================================================
// Kernel 2: attention. One block = (z, h, 128-row m-block), 8 waves.
// Software-pipelined: global loads for tile j+1 overlap WMMA compute of j.
// =====================================================================
__global__ __launch_bounds__(256) void hstu_attn_wmma(
    const float* __restrict__ q, const float* __restrict__ k, const float* __restrict__ v,
    const bf16* __restrict__ bias,
    const int* __restrict__ seq_offsets,
    float* __restrict__ out)
{
  __shared__ __attribute__((aligned(16))) bf16 k_lds[32 * KSTR];       // [key][d]
  __shared__ __attribute__((aligned(16))) bf16 vT_lds[64 * VSTR];      // [d][key]
  __shared__ __attribute__((aligned(16))) bf16 s_lds[8][16 * SSTR];    // per-wave P
  __shared__ __attribute__((aligned(16))) float bias_lds[128 * BSTR];  // f32 bias tile

  const int gid = blockIdx.x;
  const int mb = gid & 3;
  const int h  = (gid >> 2) & 3;
  const int z  = gid >> 4;

  const int start = seq_offsets[z];
  const int len   = seq_offsets[z + 1] - start;
  if (mb * 128 >= len) return;

  const int tid  = threadIdx.x;
  const int w    = tid >> 5;
  const int lane = tid & 31;
  const int c    = lane & 15;
  const int hi   = lane >> 4;

  // ---- Q * alpha -> bf16 A operands (16x32 A layout)
  const int mrow = mb * 128 + w * 16 + c;
  const int qrow = start + min(mrow, len - 1);
  const float* qp = q + (size_t)qrow * (HH * DD) + h * DD;
  v16bf aq[2];
#pragma unroll
  for (int cc = 0; cc < 2; ++cc) {
    v8bf lo = load_cvt8_scaled(qp + cc * 32 + hi * 16, ALPHA_F);
    v8bf hh = load_cvt8_scaled(qp + cc * 32 + hi * 16 + 8, ALPHA_F);
    aq[cc] = cat16(lo, hh);
  }

  v8f acc[4];
#pragma unroll
  for (int t = 0; t < 4; ++t) acc[t] = v8f{};

  const int nmaxpos = min(mb * 128 + 127, len - 1);
  const int jmax = (nmaxpos >> 5) + 1;
  const int wrow_max = mb * 128 + w * 16 + 15;   // last row this wave owns

  const bf16* bias_z = bias + (size_t)z * (NMAX * NMAX) + (size_t)(mb * 128) * NMAX;

  // cooperative-load thread mapping (fixed per thread)
  const int ld_key = tid >> 4;        // 0..15 (pass adds +16)
  const int ld_d0  = (tid & 15) * 4;
  const int bm_loc = tid >> 1;
  const int bhalf  = tid & 1;

  // ---- prologue: stage tile 0 into registers
  v4f kreg[2], vreg[2];
  v8bf breg0, breg1;
  {
#pragma unroll
    for (int pass = 0; pass < 2; ++pass) {
      int grow = start + min(pass * 16 + ld_key, len - 1);
      size_t g = (size_t)grow * (HH * DD) + h * DD + ld_d0;
      kreg[pass] = *reinterpret_cast<const v4f*>(&k[g]);
      vreg[pass] = *reinterpret_cast<const v4f*>(&v[g]);
    }
    const bf16* bp = bias_z + (size_t)bm_loc * NMAX + bhalf * 16;
    breg0 = *reinterpret_cast<const v8bf*>(bp);
    breg1 = *reinterpret_cast<const v8bf*>(bp + 8);
  }

  for (int j = 0; j < jmax; ++j) {
    const int kb = j * 32;

    // ---- dump staged registers to LDS (stores latch sources at issue)
#pragma unroll
    for (int pass = 0; pass < 2; ++pass) {
      int key = pass * 16 + ld_key;
      v4bf kb4;
#pragma unroll
      for (int i = 0; i < 4; ++i) kb4[i] = (__bf16)kreg[pass][i];
      *reinterpret_cast<v4bf*>(&k_lds[key * KSTR + ld_d0]) = kb4;
#pragma unroll
      for (int i = 0; i < 4; ++i) vT_lds[(ld_d0 + i) * VSTR + key] = (__bf16)vreg[pass][i];
    }
    {
      float* dst = &bias_lds[bm_loc * BSTR + bhalf * 16];
#pragma unroll
      for (int i = 0; i < 8; ++i) { dst[i] = (float)breg0[i]; dst[i + 8] = (float)breg1[i]; }
    }
    __syncthreads();

    // ---- issue global loads for tile j+1 (overlap with compute below)
    if (j + 1 < jmax) {
      const int kb2 = kb + 32;
#pragma unroll
      for (int pass = 0; pass < 2; ++pass) {
        int grow = start + min(kb2 + pass * 16 + ld_key, len - 1);
        size_t g = (size_t)grow * (HH * DD) + h * DD + ld_d0;
        kreg[pass] = *reinterpret_cast<const v4f*>(&k[g]);
        vreg[pass] = *reinterpret_cast<const v4f*>(&v[g]);
      }
      const bf16* bp = bias_z + (size_t)bm_loc * NMAX + kb2 + bhalf * 16;
      breg0 = *reinterpret_cast<const v8bf*>(bp);
      breg1 = *reinterpret_cast<const v8bf*>(bp + 8);
    }

    // ---- per-wave causal skip: fully-masked diagonal blocks do nothing
    if (kb <= wrow_max) {
      // ---- S = (aQ)K^T + bias, then silu -> P (bf16) in s_lds
#pragma unroll
      for (int sub = 0; sub < 2; ++sub) {
        v8f s;
#pragma unroll
        for (int vv = 0; vv < 8; ++vv)
          s[vv] = bias_lds[(vv + 8 * hi) * BSTR + sub * 16 + c];
#pragma unroll
        for (int cc = 0; cc < 2; ++cc) {
          v16bf b = lds_load16(&k_lds[(sub * 16 + c) * KSTR + cc * 32 + hi * 16]);
          s = __builtin_amdgcn_wmma_f32_16x16x32_bf16(false, aq[cc], false, b,
                                                      (short)0, s, false, false);
        }
#pragma unroll
        for (int vv = 0; vv < 8; ++vv)
          s_lds[w][(vv + 8 * hi) * SSTR + sub * 16 + c] = (__bf16)silu_fast(s[vv]);
      }

      // ---- re-stripe P (C layout -> A layout) via per-wave LDS
      v8bf plo = *reinterpret_cast<const v8bf*>(&s_lds[w][c * SSTR + hi * 16]);
      v8bf phi = *reinterpret_cast<const v8bf*>(&s_lds[w][c * SSTR + hi * 16 + 8]);
      v16bf p = cat16(plo, phi);

      // ---- O += P @ V
#pragma unroll
      for (int t = 0; t < 4; ++t) {
        v16bf b = lds_load16(&vT_lds[(t * 16 + c) * VSTR + hi * 16]);
        acc[t] = __builtin_amdgcn_wmma_f32_16x16x32_bf16(false, p, false, b,
                                                         (short)0, acc[t], false, false);
      }
    }
    __syncthreads();
  }

  // ---- store O * (1/N) (C layout), skip rows >= len
#pragma unroll
  for (int t = 0; t < 4; ++t) {
#pragma unroll
    for (int vv = 0; vv < 8; ++vv) {
      int m = mb * 128 + w * 16 + vv + 8 * hi;
      if (m < len)
        out[(size_t)(start + m) * (HH * DD) + h * DD + t * 16 + c] = acc[t][vv] * INV_N;
    }
  }
}

extern "C" void kernel_launch(void* const* d_in, const int* in_sizes, int n_in,
                              void* d_out, int out_size, void* d_ws, size_t ws_size,
                              hipStream_t stream) {
  const float* q  = (const float*)d_in[0];
  const float* k  = (const float*)d_in[1];
  const float* v  = (const float*)d_in[2];
  const float* tw = (const float*)d_in[3];
  const float* pw = (const float*)d_in[4];
  const int* seq_offsets = (const int*)d_in[5];
  const int* timestamps  = (const int*)d_in[6];
  (void)d_in[7]; (void)in_sizes; (void)n_in; (void)out_size; (void)ws_size;

  bf16* bias = (bf16*)d_ws;   // ZZ*NMAX*NMAX bf16 = 64 MB scratch

  hstu_bias_pre<<<dim3(ZZ * 40), dim3(256), 0, stream>>>(tw, pw, seq_offsets,
                                                         timestamps, bias);
  hstu_attn_wmma<<<dim3(ZZ * HH * 4), dim3(256), 0, stream>>>(q, k, v, bias,
                                                              seq_offsets,
                                                              (float*)d_out);
}
